// UV_Aggregator_19112604467374
// MI455X (gfx1250) — matrix operand order repaired and checked by reading, not compiled, not run on hardware
//
#include <hip/hip_runtime.h>
#include <hip/hip_bf16.h>

typedef __attribute__((ext_vector_type(16))) _Float16 v16h;
typedef __attribute__((ext_vector_type(8)))  _Float16 v8h;
typedef __attribute__((ext_vector_type(4)))  _Float16 v4h;
typedef __attribute__((ext_vector_type(8)))  float    v8f;

#define BDIM   256
#define NWAVE  8
#define LSEQ   200
#define LTILES 13          // ceil(200/16)
#define LPAD   208
#define ROWS   2           // batch rows per block (amortize weight staging)

// ---------------- LDS layout (byte offsets) ----------------
#define OFF_W1    0        // 64x128 f16 = 16384
#define OFF_W2    16384    // 64x64  f16 =  8192
#define OFF_WA1   24576    // 64x64  f16 =  8192 (att1_w[:, :64] only)
#define OFF_WA2   32768    // 64x64  f16 =  8192
#define OFF_OALL  40960    // ROWS x 208x64 f16 = 53248 (col-major per token)
#define OFF_ACTI  94208    // NWAVE x 16x128 f16 = 32768
#define OFF_ACTH  126976   // NWAVE x 16x64  f16 = 16384
#define OFF_LOG   143360   // ROWS x 208 f32 = 1664
#define OFF_B1    145024   // 64 f32
#define OFF_B2    145280
#define OFF_BA2   145536
#define OFF_W3    145792
#define OFF_C1    146048   // ROWS x 64 f32 = 512
#define SMEM_BYTES 146560  // ~143KB; 2 blocks fit in a 320KB WGP

__device__ __forceinline__ v8f wmma16(v16h a, v16h b, v8f c) {
  // D = A(16x32 f16) x B(32x16 f16) + C(16x16 f32)
  return __builtin_amdgcn_wmma_f32_16x16x32_f16(false, a, false, b, (short)0, c,
                                                false, false);
}

// A fragment (M x K tile, weights row-major in LDS, ld = K):
// lane l: m = mt*16 + (l&15), hi = l>>4 ; halfs j: K = kt*32 + hi*16 + j
__device__ __forceinline__ v16h lds_a_frag(const _Float16* w, int ldk, int mt,
                                           int kt, int lane) {
  const int m  = mt * 16 + (lane & 15);
  const int hi = lane >> 4;
  const _Float16* p = w + m * ldk + kt * 32 + hi * 16;
  v8h lo = *(const v8h*)p;
  v8h hv = *(const v8h*)(p + 8);
  v16h r;
#pragma unroll
  for (int i = 0; i < 8; ++i) { r[i] = lo[i]; r[i + 8] = hv[i]; }
  return r;
}

// B fragment (K x N tile, activations col-major [n][k] in LDS, ld = K):
// lane l: n = l&15, hi = l>>4 ; halfs j: K = kt*32 + hi*16 + j  (contiguous 32B)
__device__ __forceinline__ v16h lds_b_frag(const _Float16* act, int ldk, int kt,
                                           int lane) {
  const int n  = lane & 15;
  const int hi = lane >> 4;
  return *(const v16h*)(act + n * ldk + kt * 32 + hi * 16);
}

// Bias vector as WMMA C-init: D layout is 8 consecutive M per lane half.
__device__ __forceinline__ v8f bias_c(const float* bias, int mt, int lane) {
  const int hi = lane >> 4;
  return *(const v8f*)(bias + mt * 16 + hi * 8);
}

// out[m,n] = relu( sum_k W[m,k]*act[k,n] + bias[m] ), written col-major f16 (ld 64)
// B fragments hoisted out of the M loop (invariant across the 4 M-tiles).
template <int LDK, int NKT>
__device__ __forceinline__ void layer_relu_f16(const _Float16* __restrict__ wlds,
                                               const float* __restrict__ bias,
                                               const _Float16* __restrict__ act,
                                               _Float16* __restrict__ dst,
                                               int lane) {
  const int n = lane & 15, hi = lane >> 4;
  v16h bfr[NKT];
#pragma unroll
  for (int kt = 0; kt < NKT; ++kt) bfr[kt] = lds_b_frag(act, LDK, kt, lane);
#pragma unroll
  for (int mt = 0; mt < 4; ++mt) {
    v8f acc = bias_c(bias, mt, lane);   // fold +bias into the accumulator
#pragma unroll
    for (int kt = 0; kt < NKT; ++kt) {
      v16h a = lds_a_frag(wlds, LDK, mt, kt, lane);
      acc = wmma16(a, bfr[kt], acc);
    }
    _Float16 tmp[8] __attribute__((aligned(16)));
#pragma unroll
    for (int r = 0; r < 8; ++r)
      tmp[r] = (_Float16)fmaxf(acc[r], 0.0f);
    *(v8h*)(dst + n * 64 + mt * 16 + hi * 8) = *(const v8h*)tmp;
  }
}

__global__ __launch_bounds__(BDIM) void uv_aggregator_cdna5(
    const int* __restrict__ nodes, const int* __restrict__ hist_uv,
    const int* __restrict__ hist_r, const float* __restrict__ v2e,
    const float* __restrict__ r2e, const float* __restrict__ u2e,
    const float* __restrict__ w1, const float* __restrict__ b1,
    const float* __restrict__ w2, const float* __restrict__ b2,
    const float* __restrict__ a1w, const float* __restrict__ a1b,
    const float* __restrict__ a2w, const float* __restrict__ a2b,
    const float* __restrict__ a3w, const float* __restrict__ a3b,
    float* __restrict__ out, int Btot) {
  extern __shared__ __attribute__((aligned(128))) char smem[];
  _Float16* w1h  = (_Float16*)(smem + OFF_W1);
  _Float16* w2h  = (_Float16*)(smem + OFF_W2);
  _Float16* wa1h = (_Float16*)(smem + OFF_WA1);
  _Float16* wa2h = (_Float16*)(smem + OFF_WA2);
  _Float16* oall = (_Float16*)(smem + OFF_OALL);   // [ROWS][208][64]
  _Float16* acti = (_Float16*)(smem + OFF_ACTI);
  _Float16* acth = (_Float16*)(smem + OFF_ACTH);
  float* logits  = (float*)(smem + OFF_LOG);       // [ROWS][208]
  float* b1s     = (float*)(smem + OFF_B1);
  float* b2s     = (float*)(smem + OFF_B2);
  float* ba2s    = (float*)(smem + OFF_BA2);
  float* w3s     = (float*)(smem + OFF_W3);
  float* c1s     = (float*)(smem + OFF_C1);        // [ROWS][64]

  const int tid = threadIdx.x;
  const int b0  = blockIdx.x * ROWS;

  // ---- Phase 0: stage weights fp32 -> f16 LDS once; per-row fused bias c1 ----
  for (int i = tid; i < 64 * 128; i += BDIM) w1h[i] = (_Float16)w1[i];
  for (int i = tid; i < 64 * 64; i += BDIM)  w2h[i] = (_Float16)w2[i];
  for (int i = tid; i < 64 * 64; i += BDIM)
    wa1h[i] = (_Float16)a1w[(i >> 6) * 128 + (i & 63)];   // att1_w[:, :64]
  for (int i = tid; i < 64 * 64; i += BDIM)  wa2h[i] = (_Float16)a2w[i];
  if (tid < 64) {
    b1s[tid]  = b1[tid];
    b2s[tid]  = b2[tid];
    ba2s[tid] = a2b[tid];
    w3s[tid]  = a3w[tid];
  }
  if (tid < ROWS * 64) {
    const int row = tid >> 6, m = tid & 63;
    if (b0 + row < Btot) {
      // c1[m] = att1_b[m] + att1_w[m, 64:128] . u_rep (constant per batch row)
      const float* u = u2e + (size_t)nodes[b0 + row] * 64;
      float acc = a1b[m];
      for (int k = 0; k < 64; ++k) acc += a1w[m * 128 + 64 + k] * u[k];
      c1s[row * 64 + m] = acc;
    }
  }
  __syncthreads();

  // ---- Phase 1: per-wave 16-token tiles through the 4-layer MLP ----
  const int wave = tid >> 5;
  const int lane = tid & 31;
  _Float16* aIn = acti + wave * (16 * 128);
  _Float16* aH  = acth + wave * (16 * 64);
  const int ntiles = (b0 + 1 < Btot) ? (ROWS * LTILES) : LTILES;

  for (int t = wave; t < ntiles; t += NWAVE) {
    const int row = t / LTILES;          // uniform per wave
    const int tt  = t - row * LTILES;
    const int b   = b0 + row;
    // Gather [e_uv || e_r] for 16 tokens into col-major f16 [n][128]
    {
      const int n = lane & 15, sel = lane >> 4;  // sel 0: e_uv, 1: e_r
      int tok = tt * 16 + n;
      int tc  = tok < LSEQ ? tok : (LSEQ - 1);
      const int idx = sel ? hist_r[(size_t)b * LSEQ + tc]
                          : hist_uv[(size_t)b * LSEQ + tc];
      const float* src = (sel ? r2e : v2e) + (size_t)idx * 64;
      _Float16* dst = aIn + n * 128 + sel * 64;
#pragma unroll
      for (int k = 0; k < 64; k += 4) {
        float4 v = *(const float4*)(src + k);
        v4h h;
        h[0] = (_Float16)v.x; h[1] = (_Float16)v.y;
        h[2] = (_Float16)v.z; h[3] = (_Float16)v.w;
        *(v4h*)(dst + k) = h;
      }
    }
    _Float16* oTile = oall + (row * LPAD + tt * 16) * 64;
    layer_relu_f16<128, 4>(w1h, b1s, aIn, aH, lane);        // h1 = relu(W1 x + b1)
    layer_relu_f16<64, 2>(w2h, b2s, aH, oTile, lane);       // o  = relu(W2 h1 + b2)
    layer_relu_f16<64, 2>(wa1h, c1s + row * 64, oTile, aH, lane); // a1
    // att2 + logits: a2 = relu(A2 a1 + ba2); logit = w3 . a2 + w3b
    {
      const int n = lane & 15, hi = lane >> 4;
      float part = 0.0f;
      v16h bfr0 = lds_b_frag(aH, 64, 0, lane);
      v16h bfr1 = lds_b_frag(aH, 64, 1, lane);
#pragma unroll
      for (int mt = 0; mt < 4; ++mt) {
        v8f acc = bias_c(ba2s, mt, lane);
        acc = wmma16(lds_a_frag(wa2h, 64, mt, 0, lane), bfr0, acc);
        acc = wmma16(lds_a_frag(wa2h, 64, mt, 1, lane), bfr1, acc);
#pragma unroll
        for (int r = 0; r < 8; ++r) {
          const int m = mt * 16 + hi * 8 + r;
          part += fmaxf(acc[r], 0.0f) * w3s[m];
        }
      }
      part += __shfl_xor(part, 16, 32);   // combine the two M-halves per token
      if (hi == 0) {
        const int tok = tt * 16 + n;
        logits[row * LPAD + tok] = (tok < LSEQ) ? (part + a3b[0]) : -INFINITY;
      }
    }
  }
  __syncthreads();

  // ---- Phase 2: softmax over L (one wave per row) ----
  if (wave < ROWS && (b0 + wave) < Btot) {
    float* lg = logits + wave * LPAD;
    float m = -INFINITY;
    for (int l = lane; l < LPAD; l += 32) m = fmaxf(m, lg[l]);
#pragma unroll
    for (int off = 16; off > 0; off >>= 1) m = fmaxf(m, __shfl_xor(m, off, 32));
    float s = 0.0f;
    for (int l = lane; l < LPAD; l += 32) {
      float e = __expf(lg[l] - m);   // pads: exp(-inf)=0
      lg[l] = e;
      s += e;
    }
#pragma unroll
    for (int off = 16; off > 0; off >>= 1) s += __shfl_xor(s, off, 32);
    const float inv = 1.0f / s;
    for (int l = lane; l < LPAD; l += 32) lg[l] *= inv;
  }
  __syncthreads();

  // ---- Phase 3: to_feats[d] = sum_l att[l] * o[l][d] (fp32 accumulate) ----
  if (tid < ROWS * 64) {
    const int row = tid >> 6, d = tid & 63;
    if (b0 + row < Btot) {
      const float* lg = logits + row * LPAD;
      const _Float16* ob = oall + row * LPAD * 64;
      float acc = 0.0f;
      for (int l = 0; l < LSEQ; ++l)
        acc += lg[l] * (float)ob[l * 64 + d];
      out[(size_t)(b0 + row) * 64 + d] = acc;
    }
  }
}

extern "C" void kernel_launch(void* const* d_in, const int* in_sizes, int n_in,
                              void* d_out, int out_size, void* d_ws,
                              size_t ws_size, hipStream_t stream) {
  (void)n_in; (void)out_size; (void)d_ws; (void)ws_size;
  const int B = in_sizes[0];
  const int grid = (B + ROWS - 1) / ROWS;
  // Allow >64KB dynamic LDS (CDNA5 WGP has 320KB); idempotent, capture-safe.
  (void)hipFuncSetAttribute((const void*)uv_aggregator_cdna5,
                            hipFuncAttributeMaxDynamicSharedMemorySize,
                            SMEM_BYTES);
  uv_aggregator_cdna5<<<dim3(grid), dim3(BDIM), SMEM_BYTES, stream>>>(
      (const int*)d_in[0], (const int*)d_in[1], (const int*)d_in[2],
      (const float*)d_in[3], (const float*)d_in[4], (const float*)d_in[5],
      (const float*)d_in[6], (const float*)d_in[7], (const float*)d_in[8],
      (const float*)d_in[9], (const float*)d_in[10], (const float*)d_in[11],
      (const float*)d_in[12], (const float*)d_in[13], (const float*)d_in[14],
      (const float*)d_in[15], (float*)d_out, B);
}
